// MultiHeadAttention_65506841198813
// MI455X (gfx1250) — matrix operand order repaired and checked by reading, not compiled, not run on hardware
//
#include <hip/hip_runtime.h>

#define B_   4
#define S_   2048
#define D_   1024
#define H_   16
#define DK   64
#define NEG_INF_ (-1e9f)

typedef __attribute__((ext_vector_type(16))) __bf16 v16bf;
typedef __attribute__((ext_vector_type(8)))  __bf16 v8bf;
typedef __attribute__((ext_vector_type(8)))  float  v8f;
typedef __attribute__((ext_vector_type(4)))  float  v4f;
typedef __attribute__((ext_vector_type(4)))  int    v4i;

#define BK   32
#define LDT  40   // 32 + 8 bf16 pad (80B row stride: 16B aligned, conflict-free)
#define LDK  72   // 64 + 8 bf16 pad for the full-K scores tiles

// ---- CDNA5 async global->LDS path (ASYNCcnt), guarded so build never breaks.
#if defined(__gfx1250__) &&                                                    \
    __has_builtin(__builtin_amdgcn_global_load_async_to_lds_b128) &&           \
    __has_builtin(__builtin_amdgcn_s_wait_asynccnt)
#define HAVE_ASYNC_LDS 1
#else
#define HAVE_ASYNC_LDS 0
#endif

__device__ __forceinline__ void async_copy16(const __bf16* g, __bf16* l) {
#if HAVE_ASYNC_LDS
  // Builtin signature (from hipcc diagnostic): first arg is AS1 pointer to an
  // int4 vector. Do the addrspacecast first (known to work), then adjust the
  // pointee type / constness within the same address space.
  const __attribute__((address_space(1))) void* gv =
      (const __attribute__((address_space(1))) void*)g;
  __attribute__((address_space(3))) void* lv =
      (__attribute__((address_space(3))) void*)l;
  __builtin_amdgcn_global_load_async_to_lds_b128(
      (__attribute__((address_space(1))) v4i*)gv,
      (__attribute__((address_space(3))) v4i*)lv, 0, 0);
#else
  *(v8bf*)l = *(const v8bf*)g;
#endif
}

__device__ __forceinline__ void async_wait_all() {
#if HAVE_ASYNC_LDS
  __builtin_amdgcn_s_wait_asynccnt(0);
#endif
}

__device__ __forceinline__ v8f wmma_bf16(v16bf a, v16bf b, v8f c) {
  return __builtin_amdgcn_wmma_f32_16x16x32_bf16(false, a, false, b, (short)0, c,
                                                 false, false);
}

// A fragment (16x32 bf16): lane<16 holds M=lane, K in {koff+0..7, koff+16..23};
// lane>=16 holds M=lane-16, K in {koff+8..15, koff+24..31}.
template <int LD>
__device__ __forceinline__ v16bf frag_a(const __bf16* tile, int row0, int koff,
                                        int lane) {
  const int m  = lane & 15;
  const int hv = lane >> 4;
  const __bf16* p = tile + (row0 + m) * LD + koff + hv * 8;
  v8bf lo = *(const v8bf*)(p);
  v8bf hi = *(const v8bf*)(p + 16);
  return __builtin_shufflevector(lo, hi, 0, 1, 2, 3, 4, 5, 6, 7,
                                 8, 9, 10, 11, 12, 13, 14, 15);
}

// B fragment (32x16 bf16), tile stored as [N][K] (B^T rows): lane<16 holds
// column N=lane with K in {koff+0..7, koff+16..23}; lane>=16 holds N=lane-16, K in {koff+8..15, koff+24..31}.
template <int LD>
__device__ __forceinline__ v16bf frag_b(const __bf16* tile, int col0, int koff,
                                        int lane) {
  const int n  = lane & 15;
  const int hv = lane >> 4;
  const __bf16* p = tile + (col0 + n) * LD + koff + hv * 8;
  v8bf lo = *(const v8bf*)(p);
  v8bf hi = *(const v8bf*)(p + 16);
  return __builtin_shufflevector(lo, hi, 0, 1, 2, 3, 4, 5, 6, 7,
                                 8, 9, 10, 11, 12, 13, 14, 15);
}

// ---------------------------------------------------------------------------
// 1) Projection: Yh = head_split((X @ W^T + bias) * outScale) as bf16.
//    Double-buffered LDS, 128x128x32 tiles, 8 waves, wave tile 64x32.
// ---------------------------------------------------------------------------
__global__ __launch_bounds__(256) void proj_kernel(
    const float* __restrict__ X, const float* __restrict__ W,
    const float* __restrict__ bias, __bf16* __restrict__ Yh, float outScale) {
  __shared__ __bf16 As[2][128 * LDT];
  __shared__ __bf16 Bs[2][128 * LDT];
  const int t = threadIdx.x;
  const int lane = t & 31, wv = t >> 5;
  const int wm = wv >> 2, wn = wv & 3;  // 2x4 wave grid
  const int m0 = blockIdx.x * 128;
  const int n0 = blockIdx.y * 128;
  const int lr = t >> 1, lc = (t & 1) * 16;

  auto loadTile = [&](int k0, int buf) {
    const float4* ap = (const float4*)(X + (size_t)(m0 + lr) * D_ + k0 + lc);
    const float4* bp = (const float4*)(W + (size_t)(n0 + lr) * D_ + k0 + lc);
    __bf16* ad = &As[buf][lr * LDT + lc];
    __bf16* bd = &Bs[buf][lr * LDT + lc];
#pragma unroll
    for (int i = 0; i < 4; ++i) {
      float4 fa = ap[i];
      ad[i * 4 + 0] = (__bf16)fa.x; ad[i * 4 + 1] = (__bf16)fa.y;
      ad[i * 4 + 2] = (__bf16)fa.z; ad[i * 4 + 3] = (__bf16)fa.w;
      float4 fb = bp[i];
      bd[i * 4 + 0] = (__bf16)fb.x; bd[i * 4 + 1] = (__bf16)fb.y;
      bd[i * 4 + 2] = (__bf16)fb.z; bd[i * 4 + 3] = (__bf16)fb.w;
    }
  };

  const v8f vzero = {};
  v8f acc[4][2];
  for (int i = 0; i < 4; ++i)
    for (int j = 0; j < 2; ++j) acc[i][j] = vzero;

  loadTile(0, 0);
  const int NT = D_ / BK;  // 32
  for (int kt = 0; kt < NT; ++kt) {
    const int cur = kt & 1;
    __syncthreads();
    if (kt + 1 < NT) loadTile((kt + 1) * BK, cur ^ 1);
    v16bf af[4], bfm[2];
#pragma unroll
    for (int mi = 0; mi < 4; ++mi)
      af[mi] = frag_a<LDT>(As[cur], wm * 64 + mi * 16, 0, lane);
#pragma unroll
    for (int ni = 0; ni < 2; ++ni)
      bfm[ni] = frag_b<LDT>(Bs[cur], wn * 32 + ni * 16, 0, lane);
#pragma unroll
    for (int mi = 0; mi < 4; ++mi)
#pragma unroll
      for (int ni = 0; ni < 2; ++ni)
        acc[mi][ni] = wmma_bf16(af[mi], bfm[ni], acc[mi][ni]);
  }

#pragma unroll
  for (int mi = 0; mi < 4; ++mi)
#pragma unroll
    for (int ni = 0; ni < 2; ++ni) {
      const int col = n0 + wn * 32 + ni * 16 + (lane & 15);
      const int h = col >> 6, d = col & 63;
      const float bv = bias[col];
#pragma unroll
      for (int r = 0; r < 8; ++r) {
        const int row = m0 + wm * 64 + mi * 16 + (lane >> 4) * 8 + r;
        const int b = row >> 11, s = row & (S_ - 1);
        const float val = (acc[mi][ni][r] + bv) * outScale;
        Yh[(((size_t)(b * H_ + h)) * S_ + s) * DK + d] = (__bf16)val;
      }
    }
}

// ---------------------------------------------------------------------------
// 2) Scores: per (b,h): scores = Qh @ Kh^T (scale folded into Qh), masked,
//    NT-stored fp32 into attn region of d_out. K=64 fits a single LDS tile:
//    async-fill once, then two barrier-free WMMA K-steps.
// ---------------------------------------------------------------------------
__global__ __launch_bounds__(256) void scores_kernel(
    const __bf16* __restrict__ Qh, const __bf16* __restrict__ Kh,
    const int* __restrict__ mask, float* __restrict__ attn) {
  __shared__ __bf16 As[128 * LDK];
  __shared__ __bf16 Bs[128 * LDK];
  const int t = threadIdx.x;
  const int lane = t & 31, wv = t >> 5;
  const int wm = wv >> 2, wn = wv & 3;
  const int z = blockIdx.z;  // b*H + h
  const int bb = z >> 4;
  const __bf16* A  = Qh + (size_t)z * S_ * DK;
  const __bf16* Bm = Kh + (size_t)z * S_ * DK;
  const int* mk = mask + (size_t)bb * S_ * S_;
  float* outp = attn + (size_t)z * S_ * S_;
  const int m0 = blockIdx.x * 128;
  const int n0 = blockIdx.y * 128;

  // Fill the full 128x64 A and B tiles with async global->LDS copies.
  const int lr = t >> 1, lcc = (t & 1) * 32;
  {
    const __bf16* ag = A  + (size_t)(m0 + lr) * DK + lcc;
    const __bf16* bg = Bm + (size_t)(n0 + lr) * DK + lcc;
    __bf16* al = As + lr * LDK + lcc;
    __bf16* bl = Bs + lr * LDK + lcc;
#pragma unroll
    for (int i = 0; i < 4; ++i) {
      async_copy16(ag + i * 8, al + i * 8);
      async_copy16(bg + i * 8, bl + i * 8);
    }
  }
  async_wait_all();
  __syncthreads();

  const v8f vzero = {};
  v8f acc[4][2];
  for (int i = 0; i < 4; ++i)
    for (int j = 0; j < 2; ++j) acc[i][j] = vzero;

#pragma unroll
  for (int ks = 0; ks < 2; ++ks) {
    const int koff = ks * 32;
    v16bf af[4], bfm[2];
#pragma unroll
    for (int mi = 0; mi < 4; ++mi)
      af[mi] = frag_a<LDK>(As, wm * 64 + mi * 16, koff, lane);
#pragma unroll
    for (int ni = 0; ni < 2; ++ni)
      bfm[ni] = frag_b<LDK>(Bs, wn * 32 + ni * 16, koff, lane);
#pragma unroll
    for (int mi = 0; mi < 4; ++mi)
#pragma unroll
      for (int ni = 0; ni < 2; ++ni)
        acc[mi][ni] = wmma_bf16(af[mi], bfm[ni], acc[mi][ni]);
  }

#pragma unroll
  for (int mi = 0; mi < 4; ++mi)
#pragma unroll
    for (int ni = 0; ni < 2; ++ni) {
      const int col = n0 + wn * 32 + ni * 16 + (lane & 15);
#pragma unroll
      for (int r = 0; r < 8; ++r) {
        const int row = m0 + wm * 64 + mi * 16 + (lane >> 4) * 8 + r;
        const int mval = __builtin_nontemporal_load(&mk[(size_t)row * S_ + col]);
        float val = acc[mi][ni][r];
        if (mval == 0) val = NEG_INF_;
        __builtin_nontemporal_store(val, &outp[(size_t)row * S_ + col]);
      }
    }
}

// ---------------------------------------------------------------------------
// 3) Row softmax in place (NT traffic: 1 GB stream, never L2-resident).
// ---------------------------------------------------------------------------
__global__ __launch_bounds__(256) void softmax_kernel(float* __restrict__ attn) {
  float* p = attn + (size_t)blockIdx.x * S_;
  const int t = threadIdx.x, lane = t & 31, wv = t >> 5;
  v4f x0 = __builtin_nontemporal_load(((const v4f*)p) + t * 2);
  v4f x1 = __builtin_nontemporal_load(((const v4f*)p) + t * 2 + 1);
  float v[8] = {x0[0], x0[1], x0[2], x0[3], x1[0], x1[1], x1[2], x1[3]};
  float m = v[0];
#pragma unroll
  for (int i = 1; i < 8; ++i) m = fmaxf(m, v[i]);
#pragma unroll
  for (int o = 16; o > 0; o >>= 1) m = fmaxf(m, __shfl_xor(m, o, 32));
  __shared__ float red[8];
  if (lane == 0) red[wv] = m;
  __syncthreads();
  float bm = red[0];
#pragma unroll
  for (int i = 1; i < 8; ++i) bm = fmaxf(bm, red[i]);
  float s = 0.f;
#pragma unroll
  for (int i = 0; i < 8; ++i) { v[i] = __expf(v[i] - bm); s += v[i]; }
#pragma unroll
  for (int o = 16; o > 0; o >>= 1) s += __shfl_xor(s, o, 32);
  __syncthreads();
  if (lane == 0) red[wv] = s;
  __syncthreads();
  float bs = 0.f;
#pragma unroll
  for (int i = 0; i < 8; ++i) bs += red[i];
  const float inv = 1.f / bs;
  v4f y0 = {v[0] * inv, v[1] * inv, v[2] * inv, v[3] * inv};
  v4f y1 = {v[4] * inv, v[5] * inv, v[6] * inv, v[7] * inv};
  __builtin_nontemporal_store(y0, ((v4f*)p) + t * 2);
  __builtin_nontemporal_store(y1, ((v4f*)p) + t * 2 + 1);
}

// ---------------------------------------------------------------------------
// 4) attn @ V per (b,h). A = attn f32 (NT load -> bf16 into LDS), B = Vh bf16
//    transposed into LDS. Double-buffered, block 128x64x32, wave tile 32x32.
// ---------------------------------------------------------------------------
__global__ __launch_bounds__(256) void attnv_kernel(
    const float* __restrict__ attn, const __bf16* __restrict__ Vh,
    __bf16* __restrict__ Oc) {
  __shared__ __bf16 As[2][128 * LDT];
  __shared__ __bf16 Bs[2][64 * LDT];
  const int t = threadIdx.x, lane = t & 31, wv = t >> 5;
  const int wm = wv >> 1, wn = wv & 1;  // 4x2 wave grid
  const int z = blockIdx.z, bb = z >> 4, h = z & 15;
  const float* A = attn + (size_t)z * S_ * S_;
  const __bf16* V = Vh + (size_t)z * S_ * DK;
  const int m0 = blockIdx.x * 128;
  const int lr = t >> 1, lc = (t & 1) * 16;
  const int kk = t >> 3, dc = (t & 7) * 8;  // B transpose mapping

  auto loadTile = [&](int k0, int buf) {
    const v4f* ap = (const v4f*)(A + (size_t)(m0 + lr) * S_ + k0 + lc);
    __bf16* ad = &As[buf][lr * LDT + lc];
#pragma unroll
    for (int i = 0; i < 4; ++i) {
      v4f fa = __builtin_nontemporal_load(ap + i);
      ad[i * 4 + 0] = (__bf16)fa[0]; ad[i * 4 + 1] = (__bf16)fa[1];
      ad[i * 4 + 2] = (__bf16)fa[2]; ad[i * 4 + 3] = (__bf16)fa[3];
    }
    const __bf16* vp = V + (size_t)(k0 + kk) * DK + dc;
#pragma unroll
    for (int i = 0; i < 8; ++i) Bs[buf][(dc + i) * LDT + kk] = vp[i];
  };

  const v8f vzero = {};
  v8f acc[2][2];
  for (int i = 0; i < 2; ++i)
    for (int j = 0; j < 2; ++j) acc[i][j] = vzero;

  loadTile(0, 0);
  const int NT = S_ / BK;  // 64
  for (int kt = 0; kt < NT; ++kt) {
    const int cur = kt & 1;
    __syncthreads();
    if (kt + 1 < NT) loadTile((kt + 1) * BK, cur ^ 1);
    v16bf af[2], bfm[2];
#pragma unroll
    for (int mi = 0; mi < 2; ++mi)
      af[mi] = frag_a<LDT>(As[cur], wm * 32 + mi * 16, 0, lane);
#pragma unroll
    for (int ni = 0; ni < 2; ++ni)
      bfm[ni] = frag_b<LDT>(Bs[cur], wn * 32 + ni * 16, 0, lane);
#pragma unroll
    for (int mi = 0; mi < 2; ++mi)
#pragma unroll
      for (int ni = 0; ni < 2; ++ni)
        acc[mi][ni] = wmma_bf16(af[mi], bfm[ni], acc[mi][ni]);
  }

#pragma unroll
  for (int mi = 0; mi < 2; ++mi)
#pragma unroll
    for (int ni = 0; ni < 2; ++ni) {
      const int d = wn * 32 + ni * 16 + (lane & 15);
#pragma unroll
      for (int r = 0; r < 8; ++r) {
        const int s = m0 + wm * 32 + mi * 16 + (lane >> 4) * 8 + r;
        Oc[(((size_t)bb * S_ + s) * H_ + h) * DK + d] = (__bf16)acc[mi][ni][r];
      }
    }
}

// ---------------------------------------------------------------------------
// 5) Output projection: out = Oc @ Wo^T + bo (fp32). A side (bf16) uses async
//    global->LDS copies; W converted f32->bf16 in flight. Double-buffered.
// ---------------------------------------------------------------------------
__global__ __launch_bounds__(256) void outproj_kernel(
    const __bf16* __restrict__ Xc, const float* __restrict__ Wo,
    const float* __restrict__ bo, float* __restrict__ out) {
  __shared__ __bf16 As[2][128 * LDT];
  __shared__ __bf16 Bs[2][128 * LDT];
  const int t = threadIdx.x;
  const int lane = t & 31, wv = t >> 5;
  const int wm = wv >> 2, wn = wv & 3;
  const int m0 = blockIdx.x * 128;
  const int n0 = blockIdx.y * 128;
  const int lr = t >> 1, lc = (t & 1) * 16;

  auto loadTile = [&](int k0, int buf) {
    const __bf16* ag = Xc + (size_t)(m0 + lr) * D_ + k0 + lc;
    __bf16* al = &As[buf][lr * LDT + lc];
    async_copy16(ag, al);
    async_copy16(ag + 8, al + 8);
    const float4* bp = (const float4*)(Wo + (size_t)(n0 + lr) * D_ + k0 + lc);
    __bf16* bd = &Bs[buf][lr * LDT + lc];
#pragma unroll
    for (int i = 0; i < 4; ++i) {
      float4 fb = bp[i];
      bd[i * 4 + 0] = (__bf16)fb.x; bd[i * 4 + 1] = (__bf16)fb.y;
      bd[i * 4 + 2] = (__bf16)fb.z; bd[i * 4 + 3] = (__bf16)fb.w;
    }
  };

  const v8f vzero = {};
  v8f acc[4][2];
  for (int i = 0; i < 4; ++i)
    for (int j = 0; j < 2; ++j) acc[i][j] = vzero;

  loadTile(0, 0);
  const int NT = D_ / BK;  // 32
  for (int kt = 0; kt < NT; ++kt) {
    const int cur = kt & 1;
    async_wait_all();  // tile kt's async A-copies are in LDS
    __syncthreads();
    if (kt + 1 < NT) loadTile((kt + 1) * BK, cur ^ 1);
    v16bf af[4], bfm[2];
#pragma unroll
    for (int mi = 0; mi < 4; ++mi)
      af[mi] = frag_a<LDT>(As[cur], wm * 64 + mi * 16, 0, lane);
#pragma unroll
    for (int ni = 0; ni < 2; ++ni)
      bfm[ni] = frag_b<LDT>(Bs[cur], wn * 32 + ni * 16, 0, lane);
#pragma unroll
    for (int mi = 0; mi < 4; ++mi)
#pragma unroll
      for (int ni = 0; ni < 2; ++ni)
        acc[mi][ni] = wmma_bf16(af[mi], bfm[ni], acc[mi][ni]);
  }

#pragma unroll
  for (int mi = 0; mi < 4; ++mi)
#pragma unroll
    for (int ni = 0; ni < 2; ++ni) {
      const int col = n0 + wn * 32 + ni * 16 + (lane & 15);
      const float bv = bo[col];
#pragma unroll
      for (int r = 0; r < 8; ++r) {
        const int row = m0 + wm * 64 + mi * 16 + (lane >> 4) * 8 + r;
        out[(size_t)row * D_ + col] = acc[mi][ni][r] + bv;
      }
    }
}

// ---------------------------------------------------------------------------
extern "C" void kernel_launch(void* const* d_in, const int* in_sizes, int n_in,
                              void* d_out, int out_size, void* d_ws, size_t ws_size,
                              hipStream_t stream) {
  (void)in_sizes; (void)n_in; (void)out_size; (void)ws_size;
  const float* q    = (const float*)d_in[0];
  const float* k    = (const float*)d_in[1];
  const float* v    = (const float*)d_in[2];
  const int*   mask = (const int*)d_in[3];
  const float* w_q  = (const float*)d_in[4];
  const float* b_q  = (const float*)d_in[5];
  const float* w_k  = (const float*)d_in[6];
  const float* b_k  = (const float*)d_in[7];
  const float* w_v  = (const float*)d_in[8];
  const float* b_v  = (const float*)d_in[9];
  const float* w_o  = (const float*)d_in[10];
  const float* b_o  = (const float*)d_in[11];

  float* out  = (float*)d_out;
  float* attn = out + (size_t)B_ * S_ * D_;  // second tuple element

  const size_t HE = (size_t)B_ * H_ * S_ * DK;  // 8,388,608 elems (16 MB bf16)
  __bf16* qh = (__bf16*)d_ws;
  __bf16* kh = qh + HE;
  __bf16* vh = kh + HE;
  __bf16* oc = vh + HE;

  const dim3 blk(256);
  const float qscale = 0.125f;  // 1/sqrt(d_k)

  proj_kernel<<<dim3(64, 8, 1), blk, 0, stream>>>(q, w_q, b_q, qh, qscale);
  proj_kernel<<<dim3(64, 8, 1), blk, 0, stream>>>(k, w_k, b_k, kh, 1.0f);
  proj_kernel<<<dim3(64, 8, 1), blk, 0, stream>>>(v, w_v, b_v, vh, 1.0f);
  scores_kernel<<<dim3(16, 16, 64), blk, 0, stream>>>(qh, kh, mask, attn);
  softmax_kernel<<<dim3(B_ * H_ * S_, 1, 1), blk, 0, stream>>>(attn);
  attnv_kernel<<<dim3(16, 1, 64), blk, 0, stream>>>(attn, vh, oc);
  outproj_kernel<<<dim3(64, 8, 1), blk, 0, stream>>>(oc, w_o, b_o, out);
}
